// Cooord_Attn_55241869361195
// MI455X (gfx1250) — matrix operand order repaired and checked by reading, not compiled, not run on hardware
//
#include <hip/hip_runtime.h>

#define BB 2
#define CC 64
#define HH 64
#define WW 64
#define NN (HH*WW)

typedef __bf16 bf16_t;
typedef __bf16 v16bf __attribute__((ext_vector_type(16)));
typedef float  v8f   __attribute__((ext_vector_type(8)));

static __device__ inline v8f wmma_bf16(v16bf a, v16bf b, v8f c) {
  return __builtin_amdgcn_wmma_f32_16x16x32_bf16(false, a, false, b, (short)0, c, false, false);
}
static __device__ inline v8f v8f_zero() {
  v8f z;
  #pragma unroll
  for (int i = 0; i < 8; ++i) z[i] = 0.0f;
  return z;
}
static __device__ inline v16bf mk_v16(uint4 lo, uint4 hi) {
  union { uint4 u[2]; v16bf v; } u;
  u.u[0] = lo; u.u[1] = hi;
  return u.v;
}
static __device__ inline float lrelu(float x) { return x > 0.0f ? x : 0.1f * x; }

// ---------------------------------------------------------------------------
// 1) channel attention gate: mean over HW, then sigmoid(leaky(m*w+b)*w+b)
// ---------------------------------------------------------------------------
__global__ void chan_stats_k(const float* __restrict__ x, const float* __restrict__ g,
                             const float* __restrict__ lw, const float* __restrict__ lb,
                             float* __restrict__ attn) {
  int id = blockIdx.x;                    // t*B*C + b*C + c
  const float* src = (id < BB * CC) ? x : g;
  int bc = id % (BB * CC);
  const float* p = src + (size_t)bc * NN;
  __shared__ float red[256];
  float s = 0.0f;
  for (int i = threadIdx.x; i < NN; i += 256) s += p[i];
  red[threadIdx.x] = s;
  __syncthreads();
  for (int off = 128; off > 0; off >>= 1) {
    if ((int)threadIdx.x < off) red[threadIdx.x] += red[threadIdx.x + off];
    __syncthreads();
  }
  if (threadIdx.x == 0) {
    float mean = red[0] * (1.0f / NN);
    float w = lw[0], b = lb[0];
    float h1 = lrelu(mean * w + b);
    float s2 = h1 * w + b;
    attn[id] = 1.0f / (1.0f + __expf(-s2));
  }
}

// ---------------------------------------------------------------------------
// 2) gated coord-conv -> bf16 (N,C). Block = one pixel n; thread = out chan o.
//    Input taps are wave-uniform (scalar loads); stores are contiguous 128B.
// ---------------------------------------------------------------------------
__global__ __launch_bounds__(64) void gated_conv_k(
    const float* __restrict__ x, const float* __restrict__ g,
    const float* __restrict__ cw, const float* __restrict__ attn,
    bf16_t* __restrict__ xg_t, bf16_t* __restrict__ gg_t) {
  int n = blockIdx.x;
  int h = n >> 6, w = n & 63;
  int t = blockIdx.y & 1, b = blockIdx.y >> 1;
  int o = threadIdx.x;
  const float* in = t ? g : x;
  const float* wrow = cw + (size_t)o * 66 * 9;
  float acc = 0.0f;
  for (int ci = 0; ci < CC; ++ci) {
    const float* ip = in + (size_t)((b * CC + ci) * HH) * WW;
    const float* wp = wrow + ci * 9;
    #pragma unroll
    for (int dh = -1; dh <= 1; ++dh) {
      int hh = h + dh; if ((unsigned)hh >= HH) continue;
      const float* rp = ip + hh * WW;
      #pragma unroll
      for (int dw = -1; dw <= 1; ++dw) {
        int ww_ = w + dw; if ((unsigned)ww_ >= WW) continue;
        acc += rp[ww_] * wp[(dh + 1) * 3 + (dw + 1)];
      }
    }
  }
  const float* wx = wrow + 64 * 9;   // xx channel (varies along W)
  const float* wy = wrow + 65 * 9;   // yy channel (varies along H)
  #pragma unroll
  for (int dh = -1; dh <= 1; ++dh) {
    int hh = h + dh; if ((unsigned)hh >= HH) continue;
    float yv = hh * (2.0f / (HH - 1)) - 1.0f;
    #pragma unroll
    for (int dw = -1; dw <= 1; ++dw) {
      int ww_ = w + dw; if ((unsigned)ww_ >= WW) continue;
      float xv = ww_ * (2.0f / (WW - 1)) - 1.0f;
      acc += xv * wx[(dh + 1) * 3 + (dw + 1)];
      acc += yv * wy[(dh + 1) * 3 + (dw + 1)];
    }
  }
  float a = attn[t * BB * CC + b * CC + o];
  bf16_t* outp = t ? gg_t : xg_t;
  outp[((size_t)b * NN + n) * CC + o] = (bf16_t)(a * acc);
}

// ---------------------------------------------------------------------------
// 3) 1x1 conv projections as WMMA GEMM: Y[o,n] = sum_c W[o,c] X[c,n] + b[o]
//    One wave computes 64(o) x 16(n). Q/K out (N,C) bf16; V out (C,N) bf16.
// ---------------------------------------------------------------------------
__global__ __launch_bounds__(32) void proj_k(
    const bf16_t* __restrict__ xg_t, const bf16_t* __restrict__ gg_t,
    const float* __restrict__ xqw, const float* __restrict__ xqb,
    const float* __restrict__ xkw, const float* __restrict__ xkb,
    const float* __restrict__ xvw, const float* __restrict__ xvb,
    const float* __restrict__ gqw, const float* __restrict__ gqb,
    const float* __restrict__ gkw, const float* __restrict__ gkb,
    bf16_t* __restrict__ Qx, bf16_t* __restrict__ Kx, bf16_t* __restrict__ Vx,
    bf16_t* __restrict__ Qg, bf16_t* __restrict__ Kg) {
  int lane = threadIdx.x;
  int half = lane >> 4, col = lane & 15;
  int n0 = blockIdx.x * 16;
  int pb = blockIdx.y;
  int p = pb % 5, b = pb / 5;
  const bf16_t* src = ((p < 3) ? xg_t : gg_t) + (size_t)b * NN * CC;
  const float* Wm; const float* bias; bf16_t* dst;
  switch (p) {
    case 0: Wm = xqw; bias = xqb; dst = Qx; break;
    case 1: Wm = xkw; bias = xkb; dst = Kx; break;
    case 2: Wm = xvw; bias = xvb; dst = Vx; break;
    case 3: Wm = gqw; bias = gqb; dst = Qg; break;
    default: Wm = gkw; bias = gkb; dst = Kg; break;
  }
  dst += (size_t)b * NN * CC;

  // B fragments (activation tile (N,C) bf16): contiguous 32B per fragment
  const uint4* sp4 = (const uint4*)(src + (size_t)(n0 + col) * CC);
  v16bf Bf[2];
  #pragma unroll
  for (int j = 0; j < 2; ++j)
    Bf[j] = mk_v16(sp4[4 * j + 2 * half], sp4[4 * j + 2 * half + 1]);

  v8f accT[4];
  #pragma unroll
  for (int ot = 0; ot < 4; ++ot) {
    v8f acc = v8f_zero();
    #pragma unroll
    for (int j = 0; j < 2; ++j) {
      v16bf Af;   // weights, f32 -> bf16 (contiguous runs; compiler vectorizes)
      #pragma unroll
      for (int e = 0; e < 16; ++e) {
        int k = 16 * half + e;
        int o = ot * 16 + col;
        Af[e] = (bf16_t)Wm[o * 64 + j * 32 + k];
      }
      acc = wmma_bf16(Af, Bf[j], acc);
    }
    accT[ot] = acc;
  }

  if (p == 2) {                         // V: (C,N) scatter (columns of the map)
    #pragma unroll
    for (int ot = 0; ot < 4; ++ot)
      #pragma unroll
      for (int r = 0; r < 8; ++r) {
        int o = ot * 16 + 8 * half + r;
        dst[(size_t)o * NN + n0 + col] = (bf16_t)(accT[ot][r] + bias[o]);
      }
  } else {                              // Q/K: (N,C), 8 contiguous bf16 -> b128
    #pragma unroll
    for (int ot = 0; ot < 4; ++ot) {
      union { bf16_t h[8]; uint4 u; } pk;
      #pragma unroll
      for (int r = 0; r < 8; ++r) {
        int o = ot * 16 + 8 * half + r;
        pk.h[r] = (bf16_t)(accT[ot][r] + bias[o]);
      }
      *(uint4*)(dst + (size_t)(n0 + col) * CC + ot * 16 + 8 * half) = pk.u;
    }
  }
}

// ---------------------------------------------------------------------------
// 4) fused flash attention with async global->LDS K/V tiles (double-buffered)
//    Q/K in (N,C) bf16, V in (C,N) bf16. Wave = 16 query rows.
// ---------------------------------------------------------------------------
static __device__ inline void async_tiles(const bf16_t* Kp, const bf16_t* Vt,
                                          bf16_t* kbuf, bf16_t* vbuf, int m0, int tid) {
  // K tile: 32 rows x 64 bf16, contiguous 4KB. offset: applies to LDS+global.
  {
    unsigned lds = (unsigned)(size_t)kbuf + tid * 16;
    unsigned long long ga = (unsigned long long)(size_t)(Kp + (size_t)m0 * CC) + tid * 16;
    asm volatile("global_load_async_to_lds_b128 %0, %1, off\n\t"
                 "global_load_async_to_lds_b128 %0, %1, off offset:2048"
                 :: "v"(lds), "v"(ga) : "memory");
  }
  // V tile (transposed): 64 rows (c) x 32 bf16 (m), global stride NN per row.
  {
    int c = tid >> 2, s = tid & 3;
    unsigned lds0 = (unsigned)(size_t)vbuf + (c * 32 + s * 8) * 2;
    unsigned long long ga0 =
        (unsigned long long)(size_t)(Vt + (size_t)c * NN + m0 + s * 8);
    asm volatile("global_load_async_to_lds_b128 %0, %1, off"
                 :: "v"(lds0), "v"(ga0) : "memory");
    unsigned lds1 = lds0 + 32 * 32 * 2;
    unsigned long long ga1 = ga0 + (unsigned long long)32 * NN * 2;
    asm volatile("global_load_async_to_lds_b128 %0, %1, off"
                 :: "v"(lds1), "v"(ga1) : "memory");
  }
}

__global__ __launch_bounds__(128) void flash_k(
    const bf16_t* __restrict__ Qx, const bf16_t* __restrict__ Kx,
    const bf16_t* __restrict__ Vx, const bf16_t* __restrict__ Qg,
    const bf16_t* __restrict__ Kg,
    float* __restrict__ xout, float* __restrict__ gout) {
  __shared__ __align__(16) bf16_t K_lds[2][32 * 64];
  __shared__ __align__(16) bf16_t V_lds[2][64 * 32];   // transposed (c, m)
  __shared__ __align__(16) bf16_t P_lds[4 * 16 * 32];  // per-wave 16x32 P tile

  int a = blockIdx.y & 1;
  int b = blockIdx.y >> 1;
  const bf16_t* Q  = (a ? Qg : Qx) + (size_t)b * NN * CC;
  const bf16_t* Kp = (a ? Kg : Kx) + (size_t)b * NN * CC;
  const bf16_t* Vt = Vx + (size_t)b * NN * CC;         // (C,N)
  float* Out = (a ? gout : xout) + (size_t)b * CC * NN;

  int tid = threadIdx.x;
  int wv = tid >> 5;
  int lane = tid & 31;
  int half = lane >> 4, col = lane & 15;
  int qbase = blockIdx.x * 64 + wv * 16;

  // Q fragments (A-layout): contiguous 32B per k-chunk
  const uint4* qp4 = (const uint4*)(Q + (size_t)(qbase + col) * CC);
  v16bf Qf[2];
  #pragma unroll
  for (int j = 0; j < 2; ++j)
    Qf[j] = mk_v16(qp4[4 * j + 2 * half], qp4[4 * j + 2 * half + 1]);

  float m_run[8], l_run[8];
  v8f O[4];
  #pragma unroll
  for (int r = 0; r < 8; ++r) { m_run[r] = -1e30f; l_run[r] = 0.0f; }
  #pragma unroll
  for (int t = 0; t < 4; ++t) O[t] = v8f_zero();

  async_tiles(Kp, Vt, K_lds[0], V_lds[0], 0, tid);

  for (int it = 0; it < NN / 32; ++it) {
    int m0 = it * 32;
    int buf = it & 1;
    bool hasnext = (it + 1) < (NN / 32);
    if (hasnext) {
      async_tiles(Kp, Vt, K_lds[buf ^ 1], V_lds[buf ^ 1], m0 + 32, tid);
      asm volatile("s_wait_asynccnt 0x4" ::: "memory");   // drain current tile only
    } else {
      asm volatile("s_wait_asynccnt 0x0" ::: "memory");
    }
    __syncthreads();

    const bf16_t* kb = K_lds[buf];
    const bf16_t* vb = V_lds[buf];

    // S tiles: 16 q-rows x 16 m-cols, two tiles (mj = 0, 16)
    v8f S[2];
    #pragma unroll
    for (int mj = 0; mj < 2; ++mj) {
      v8f acc = v8f_zero();
      const uint4* kp4 = (const uint4*)(kb + (mj * 16 + col) * 64);
      #pragma unroll
      for (int j = 0; j < 2; ++j) {
        v16bf Bf = mk_v16(kp4[4 * j + 2 * half], kp4[4 * j + 2 * half + 1]);
        acc = wmma_bf16(Qf[j], Bf, acc);
      }
      S[mj] = acc;
    }

    // online softmax: VGPR r <-> q-row r + 8*half; reduce over 16-lane halves
    float P0[8], P1[8];
    #pragma unroll
    for (int r = 0; r < 8; ++r) {
      float tm = fmaxf(S[0][r], S[1][r]);
      for (int msk = 1; msk < 16; msk <<= 1) tm = fmaxf(tm, __shfl_xor(tm, msk, 32));
      float nm = fmaxf(m_run[r], tm);
      float sc = __expf(m_run[r] - nm);
      float p0 = __expf(S[0][r] - nm);
      float p1 = __expf(S[1][r] - nm);
      float rs = p0 + p1;
      for (int msk = 1; msk < 16; msk <<= 1) rs += __shfl_xor(rs, msk, 32);
      l_run[r] = l_run[r] * sc + rs;
      m_run[r] = nm;
      P0[r] = p0; P1[r] = p1;
      #pragma unroll
      for (int t = 0; t < 4; ++t) O[t][r] *= sc;
    }

    // stage P: D-layout -> row-major LDS (transpose), read back as A fragment
    int pbase = wv * 512;
    #pragma unroll
    for (int r = 0; r < 8; ++r) {
      int row = r + 8 * half;
      P_lds[pbase + row * 32 + col]      = (bf16_t)P0[r];
      P_lds[pbase + row * 32 + 16 + col] = (bf16_t)P1[r];
    }
    const uint4* pp4 = (const uint4*)(P_lds + pbase + col * 32);
    v16bf Pf = mk_v16(pp4[2 * half], pp4[2 * half + 1]);

    // O += P (16x32) x V (32x16 per c-chunk); V fragment contiguous in (c,m)
    #pragma unroll
    for (int t = 0; t < 4; ++t) {
      const uint4* vp4 = (const uint4*)(vb + (t * 16 + col) * 32);
      v16bf Bf = mk_v16(vp4[2 * half], vp4[2 * half + 1]);
      O[t] = wmma_bf16(Pf, Bf, O[t]);
    }
    __syncthreads();
  }

  // normalize and write (C,N) f32
  #pragma unroll
  for (int r = 0; r < 8; ++r) {
    float inv = 1.0f / l_run[r];
    int n = qbase + r + 8 * half;
    #pragma unroll
    for (int t = 0; t < 4; ++t)
      Out[(size_t)(t * 16 + col) * NN + n] = O[t][r] * inv;
  }
}

// ---------------------------------------------------------------------------
// 5) out = gamma * x_out + alpha * guide_out
// ---------------------------------------------------------------------------
__global__ void combine_k(const float* __restrict__ xout, const float* __restrict__ gout,
                          const float* __restrict__ gamma, const float* __restrict__ alpha,
                          float* __restrict__ outbuf) {
  int i = blockIdx.x * blockDim.x + threadIdx.x;
  if (i < BB * CC * NN) outbuf[i] = gamma[0] * xout[i] + alpha[0] * gout[i];
}

// ---------------------------------------------------------------------------
// 6) y = conv3x3(leaky(in), w) + b
// ---------------------------------------------------------------------------
__global__ void conv3x3_k(const float* __restrict__ in, const float* __restrict__ wts,
                          const float* __restrict__ bias, float* __restrict__ out) {
  int idx = blockIdx.x;
  int b = idx / CC, o = idx % CC;
  int h = blockIdx.y, w = threadIdx.x;
  float acc = bias[o];
  const float* wrow = wts + (size_t)o * CC * 9;
  for (int ci = 0; ci < CC; ++ci) {
    const float* ip = in + (size_t)((b * CC + ci) * HH) * WW;
    const float* wp = wrow + ci * 9;
    #pragma unroll
    for (int dh = -1; dh <= 1; ++dh) {
      int hh = h + dh; if ((unsigned)hh >= HH) continue;
      const float* rp = ip + hh * WW;
      #pragma unroll
      for (int dw = -1; dw <= 1; ++dw) {
        int ww_ = w + dw; if ((unsigned)ww_ >= WW) continue;
        acc += lrelu(rp[ww_]) * wp[(dh + 1) * 3 + (dw + 1)];
      }
    }
  }
  out[(size_t)((b * CC + o) * HH + h) * WW + w] = acc;
}

// ---------------------------------------------------------------------------
// 7) final = branch + (conv1x1(out, sc_w) + sc_b) * guide_out
// ---------------------------------------------------------------------------
__global__ void final_k(const float* __restrict__ branch, const float* __restrict__ outbuf,
                        const float* __restrict__ gout, const float* __restrict__ scw,
                        const float* __restrict__ scb, float* __restrict__ out) {
  int n = blockIdx.x * blockDim.x + threadIdx.x;
  int bc = blockIdx.y;
  int b = bc / CC, c = bc % CC;
  const float* ob = outbuf + (size_t)b * CC * NN;
  float acc = scb[c];
  const float* wr = scw + c * CC;
  for (int c2 = 0; c2 < CC; ++c2) acc += wr[c2] * ob[(size_t)c2 * NN + n];
  size_t i = (size_t)(b * CC + c) * NN + n;
  out[i] = branch[i] + acc * gout[i];
}

// ---------------------------------------------------------------------------
extern "C" void kernel_launch(void* const* d_in, const int* in_sizes, int n_in,
                              void* d_out, int out_size, void* d_ws, size_t ws_size,
                              hipStream_t stream) {
  (void)in_sizes; (void)n_in; (void)out_size; (void)ws_size;
  const float* x      = (const float*)d_in[0];
  const float* guide  = (const float*)d_in[1];
  const float* lin_w  = (const float*)d_in[2];
  const float* lin_b  = (const float*)d_in[3];
  const float* coordw = (const float*)d_in[4];
  const float* xq_w = (const float*)d_in[5];  const float* xq_b = (const float*)d_in[6];
  const float* xk_w = (const float*)d_in[7];  const float* xk_b = (const float*)d_in[8];
  const float* xv_w = (const float*)d_in[9];  const float* xv_b = (const float*)d_in[10];
  const float* gq_w = (const float*)d_in[11]; const float* gq_b = (const float*)d_in[12];
  const float* gk_w = (const float*)d_in[13]; const float* gk_b = (const float*)d_in[14];
  const float* gamma = (const float*)d_in[15];
  const float* alpha = (const float*)d_in[16];
  const float* c1_w = (const float*)d_in[17]; const float* c1_b = (const float*)d_in[18];
  const float* c2_w = (const float*)d_in[19]; const float* c2_b = (const float*)d_in[20];
  const float* sc_w = (const float*)d_in[21]; const float* sc_b = (const float*)d_in[22];
  float* out = (float*)d_out;

  char* ws = (char*)d_ws;
  size_t off = 0;
  auto alloc = [&](size_t bytes) -> char* {
    char* p = ws + off;
    off += (bytes + 255) & ~(size_t)255;
    return p;
  };
  const size_t fmap = (size_t)BB * CC * NN;         // 524288 elements
  float*  attn_s = (float*)alloc(2 * BB * CC * sizeof(float));
  bf16_t* xg_t   = (bf16_t*)alloc(fmap * sizeof(bf16_t));   // (B,N,C)
  bf16_t* gg_t   = (bf16_t*)alloc(fmap * sizeof(bf16_t));
  bf16_t* Qx     = (bf16_t*)alloc(fmap * sizeof(bf16_t));   // (B,N,C)
  bf16_t* Kx     = (bf16_t*)alloc(fmap * sizeof(bf16_t));   // (B,N,C)
  bf16_t* Vx     = (bf16_t*)alloc(fmap * sizeof(bf16_t));   // (B,C,N)
  bf16_t* Qg     = (bf16_t*)alloc(fmap * sizeof(bf16_t));
  bf16_t* Kg     = (bf16_t*)alloc(fmap * sizeof(bf16_t));
  float*  xout   = (float*)alloc(fmap * sizeof(float));
  float*  gout   = (float*)alloc(fmap * sizeof(float));
  float*  outbuf = (float*)alloc(fmap * sizeof(float));
  float*  t1     = (float*)alloc(fmap * sizeof(float));
  float*  branch = (float*)alloc(fmap * sizeof(float));

  chan_stats_k<<<2 * BB * CC, 256, 0, stream>>>(x, guide, lin_w, lin_b, attn_s);
  gated_conv_k<<<dim3(NN, 2 * BB), 64, 0, stream>>>(x, guide, coordw, attn_s, xg_t, gg_t);
  proj_k<<<dim3(NN / 16, 5 * BB), 32, 0, stream>>>(xg_t, gg_t,
      xq_w, xq_b, xk_w, xk_b, xv_w, xv_b, gq_w, gq_b, gk_w, gk_b,
      Qx, Kx, Vx, Qg, Kg);
  flash_k<<<dim3(NN / 64, 2 * BB), 128, 0, stream>>>(Qx, Kx, Vx, Qg, Kg, xout, gout);
  combine_k<<<(int)((fmap + 255) / 256), 256, 0, stream>>>(xout, gout, gamma, alpha, outbuf);
  conv3x3_k<<<dim3(BB * CC, HH), WW, 0, stream>>>(outbuf, c1_w, c1_b, t1);
  conv3x3_k<<<dim3(BB * CC, HH), WW, 0, stream>>>(t1, c2_w, c2_b, branch);
  final_k<<<dim3(NN / 256, BB * CC), 256, 0, stream>>>(branch, outbuf, gout, sc_w, sc_b, out);
}
